// EncoderLayer_67018669686884
// MI455X (gfx1250) — matrix operand order repaired and checked by reading, not compiled, not run on hardware
//
#include <hip/hip_runtime.h>

typedef __bf16 bf16_t;
typedef __attribute__((ext_vector_type(8)))  __bf16 v8bf;
typedef __attribute__((ext_vector_type(16))) __bf16 v16bf;
typedef __attribute__((ext_vector_type(8)))  float  v8f;

#define WMMA_BF16(a, b, c) \
  __builtin_amdgcn_wmma_f32_16x16x32_bf16(false, (a), false, (b), (short)0, (c), false, false)

static __device__ __forceinline__ v16bf combine16(v8bf lo, v8bf hi) {
  v16bf r;
#pragma unroll
  for (int i = 0; i < 8; ++i) { r[i] = lo[i]; r[i + 8] = hi[i]; }
  return r;
}

// Async global->LDS copy of 16 bytes per lane (CDNA5, ASYNCcnt-tracked).
static __device__ __forceinline__ void async_load_b128(void* lds_ptr, const void* gptr) {
  const unsigned lds_off = (unsigned)(size_t)lds_ptr;  // low 32 bits = LDS offset
  asm volatile("global_load_async_to_lds_b128 %0, %1, off"
               :: "v"(lds_off), "v"(gptr) : "memory");
}
static __device__ __forceinline__ void wait_async0() {
  asm volatile("s_wait_asynccnt 0x0" ::: "memory");
}

// ---------------------------------------------------------------------------
// f32 -> bf16 elementwise conversion
// ---------------------------------------------------------------------------
__global__ __launch_bounds__(256) void convert_f32_bf16_kernel(
    const float* __restrict__ in, bf16_t* __restrict__ out, size_t n)
{
  size_t i = (size_t)blockIdx.x * blockDim.x + threadIdx.x;
  const size_t stride = (size_t)gridDim.x * blockDim.x;
  for (; i < n; i += stride) out[i] = (bf16_t)in[i];
}

// ---------------------------------------------------------------------------
// bf16 GEMM with double-buffered async global->LDS tile pipeline.
// MODE 0: out = acc
// MODE 1: out = gate * (acc + bias[col])
// MODE 2: out = gate * acc + addend
// BM=128, BN=128, BK=32; 256 threads = 8 wave32; wave tile 32x64 (2x4 wmma)
// ---------------------------------------------------------------------------
template <int MODE>
__global__ __launch_bounds__(256) void gemm_bf16_kernel(
    const bf16_t* __restrict__ A, const bf16_t* __restrict__ W,
    const float* __restrict__ bias, const float* __restrict__ gate,
    const bf16_t* __restrict__ addend, bf16_t* __restrict__ out,
    int M, int N, int K)
{
  __shared__ bf16_t lsA[2][128 * 40];   // [m][k], k-pad 8
  __shared__ bf16_t lsB[2][32 * 136];   // [k][n], n-pad 8
  const int nb = N >> 7;
  const int tm = blockIdx.x / nb, tn = blockIdx.x % nb;
  const int m0 = tm << 7, n0 = tn << 7;
  const int t = threadIdx.x;
  const int lane = t & 31, wave = t >> 5;
  const int wm = wave & 3, wn = wave >> 2;     // 4 x 2 wave grid
  const int lm = lane & 15, khf = lane >> 4;

  // Fixed per-thread tile-load coordinates.
  const int ar = t >> 2, ac = (t & 3) * 8;     // A: rows 0..63 (+64 second pass)
  const int br = t >> 4, bc = (t & 15) * 8;    // B: rows 0..15 (+16 second pass)

  v8f acc[2][4] = {};

  auto issue = [&](int k0, int buf) {
    const bf16_t* a0 = A + (size_t)(m0 + ar) * K + k0 + ac;
    async_load_b128(&lsA[buf][ar * 40 + ac], a0);
    async_load_b128(&lsA[buf][(ar + 64) * 40 + ac], a0 + (size_t)64 * K);
    const bf16_t* w0 = W + (size_t)(k0 + br) * N + n0 + bc;
    async_load_b128(&lsB[buf][br * 136 + bc], w0);
    async_load_b128(&lsB[buf][(br + 16) * 136 + bc], w0 + (size_t)16 * N);
  };

  issue(0, 0);
  wait_async0();
  __syncthreads();

  for (int k0 = 0; k0 < K; k0 += 32) {
    const int cur = (k0 >> 5) & 1;
    if (k0 + 32 < K) issue(k0 + 32, cur ^ 1);

    v16bf af[2], bfv[4];
#pragma unroll
    for (int i = 0; i < 2; ++i) {
      const int m = wm * 32 + i * 16 + lm;
      af[i] = combine16(*(const v8bf*)&lsA[cur][m * 40 + khf * 8],
                        *(const v8bf*)&lsA[cur][m * 40 + 16 + khf * 8]);
    }
#pragma unroll
    for (int j = 0; j < 4; ++j) {
      const int n = wn * 64 + j * 16;
#pragma unroll
      for (int s = 0; s < 16; ++s)
        bfv[j][s] = lsB[cur][((s >> 3) * 16 + khf * 8 + (s & 7)) * 136 + n + lm];
    }
#pragma unroll
    for (int i = 0; i < 2; ++i)
#pragma unroll
      for (int j = 0; j < 4; ++j)
        acc[i][j] = WMMA_BF16(af[i], bfv[j], acc[i][j]);

    wait_async0();
    __syncthreads();
  }

  // Epilogue. C layout: vgpr r -> row = (lane>>4)*8 + r, col = lane&15.
  const float g = (MODE == 0) ? 1.0f : gate[0];
#pragma unroll
  for (int i = 0; i < 2; ++i) {
#pragma unroll
    for (int j = 0; j < 4; ++j) {
      const int col = n0 + wn * 64 + j * 16 + lm;
      const float bv = (MODE == 1) ? bias[col] : 0.0f;
#pragma unroll
      for (int r = 0; r < 8; ++r) {
        const int row = m0 + wm * 32 + i * 16 + khf * 8 + r;
        const size_t off = (size_t)row * N + col;
        float v = acc[i][j][r];
        if (MODE == 1) v = (v + bv) * g;
        if (MODE == 2) v = v * g + (float)addend[off];
        out[off] = (bf16_t)v;
      }
    }
  }
}

// ---------------------------------------------------------------------------
// Transpose [B,S,H,DH] bf16 -> [B,H,DH,S] bf16 (for K^T)
// ---------------------------------------------------------------------------
__global__ __launch_bounds__(256) void transpose_sdh_kernel(
    const bf16_t* __restrict__ in, bf16_t* __restrict__ out)
{
  const int S = 1024, Hn = 8, DH = 128;
  __shared__ bf16_t tile[32][33];
  int r = blockIdx.x;
  const int dt = r & 3;  r >>= 2;    // DH/32 = 4
  const int st = r & 31; r >>= 5;    // S/32 = 32
  const int h = r & 7;   r >>= 3;
  const int b = r;
  const int t = threadIdx.x;
  const int rr = t >> 3;             // 0..31
  const int c4 = t & 7;
#pragma unroll
  for (int q = 0; q < 4; ++q) {
    const int d = c4 * 4 + q;
    tile[rr][d] = in[(((size_t)b * S + st * 32 + rr) * Hn + h) * DH + dt * 32 + d];
  }
  __syncthreads();
#pragma unroll
  for (int q = 0; q < 4; ++q) {
    const int sidx = c4 * 4 + q;
    out[(((size_t)b * Hn + h) * DH + dt * 32 + rr) * S + st * 32 + sidx] = tile[sidx][rr];
  }
}

// ---------------------------------------------------------------------------
// Flash attention: qh [B,S,H,DH], kt [B,H,DH,S], vh [B,S,H,DH] -> ctx [B,S,H,DH]
// block = (b,h,qtile of 128 rows); 8 waves, 16 q-rows per wave; k-tiles of 64;
// double-buffered async K^T/V tiles.
// ---------------------------------------------------------------------------
__global__ __launch_bounds__(256) void attn_kernel(
    const bf16_t* __restrict__ qh, const bf16_t* __restrict__ kt,
    const bf16_t* __restrict__ vh, bf16_t* __restrict__ ctx)
{
  const int S = 1024, Hn = 8, DH = 128;
  __shared__ bf16_t lsK[2][128 * 72];     // [d=128][s=64 +8 pad]
  __shared__ bf16_t lsV[2][64 * 136];     // [s=64][d=128 +8 pad]
  __shared__ bf16_t lsP[8 * 16 * 72];     // per-wave P stage [16][64 +8 pad]
  const int qt = blockIdx.x & 7;
  const int bh = blockIdx.x >> 3;
  const int b = bh >> 3, h = bh & 7;
  const int t = threadIdx.x, lane = t & 31, wave = t >> 5;
  const int lm = lane & 15, khf = lane >> 4;
  const int q0 = qt * 128 + wave * 16;

  const bf16_t* ktb = kt + ((size_t)b * Hn + h) * DH * S;

  auto issue = [&](int j, int buf) {
#pragma unroll
    for (int p = 0; p < 4; ++p) {
      const int idx = p * 256 + t;
      { const int row = idx >> 3, ch = (idx & 7) * 8;   // K^T tile: 128 d x 64 s
        async_load_b128(&lsK[buf][row * 72 + ch], ktb + (size_t)row * S + j * 64 + ch); }
      { const int row = idx >> 4, ch = (idx & 15) * 8;  // V tile: 64 s x 128 d
        async_load_b128(&lsV[buf][row * 136 + ch],
                        vh + (((size_t)b * S + j * 64 + row) * Hn + h) * DH + ch); }
    }
  };

  // Q fragments (A-layout): lane holds row q0+lm, split-K runs of 8 in d.
  v16bf qf[4];
  const bf16_t* qbase = qh + (((size_t)b * S + q0 + lm) * Hn + h) * DH;
#pragma unroll
  for (int dc = 0; dc < 4; ++dc)
    qf[dc] = combine16(*(const v8bf*)(qbase + dc * 32 + khf * 8),
                       *(const v8bf*)(qbase + dc * 32 + 16 + khf * 8));

  v8f acco[8] = {};
  float mstat[8], lstat[8];
#pragma unroll
  for (int i = 0; i < 8; ++i) { mstat[i] = -1e30f; lstat[i] = 0.0f; }
  const float rscale = 0.08838834764831845f;  // 1/sqrt(128)
  bf16_t* pp = &lsP[wave * 16 * 72];

  issue(0, 0);
  wait_async0();
  __syncthreads();

  for (int j = 0; j < 16; ++j) {
    const int cur = j & 1;
    if (j + 1 < 16) issue(j + 1, cur ^ 1);

    // scores = Q @ K^T  (B frag: lane = s column, d in split-K runs of 8)
    v8f accs[4];
#pragma unroll
    for (int nt = 0; nt < 4; ++nt) {
      v8f c = {};
#pragma unroll
      for (int dc = 0; dc < 4; ++dc) {
        v16bf bfr;
#pragma unroll
        for (int s = 0; s < 16; ++s)
          bfr[s] = lsK[cur][(dc * 32 + (s >> 3) * 16 + khf * 8 + (s & 7)) * 72 + nt * 16 + lm];
        c = WMMA_BF16(qf[dc], bfr, c);
      }
      accs[nt] = c;
    }
    // online softmax per row (row = khf*8 + i, spread over 16 lanes of a half)
#pragma unroll
    for (int i = 0; i < 8; ++i) {
      float mx = -1e30f;
#pragma unroll
      for (int nt = 0; nt < 4; ++nt) {
        const float v = accs[nt][i] * rscale;
        accs[nt][i] = v;
        mx = fmaxf(mx, v);
      }
#pragma unroll
      for (int msk = 1; msk < 16; msk <<= 1) mx = fmaxf(mx, __shfl_xor(mx, msk, 32));
      const float mnew = fmaxf(mstat[i], mx);
      const float alpha = __expf(mstat[i] - mnew);
      float srow = 0.0f;
#pragma unroll
      for (int nt = 0; nt < 4; ++nt) {
        const float pv = __expf(accs[nt][i] - mnew);
        accs[nt][i] = pv;
        srow += pv;
      }
#pragma unroll
      for (int msk = 1; msk < 16; msk <<= 1) srow += __shfl_xor(srow, msk, 32);
      lstat[i] = lstat[i] * alpha + srow;
      mstat[i] = mnew;
#pragma unroll
      for (int dt = 0; dt < 8; ++dt) acco[dt][i] *= alpha;
      const int prow = khf * 8 + i;
#pragma unroll
      for (int nt = 0; nt < 4; ++nt)
        pp[prow * 72 + nt * 16 + lm] = (bf16_t)accs[nt][i];
    }
    // Same-wave LDS ops are processed in order: no barrier needed between the
    // P stage writes (private region) and the fragment reads below.
#pragma unroll
    for (int kc = 0; kc < 2; ++kc) {
      v16bf pa = combine16(*(const v8bf*)&pp[lm * 72 + kc * 32 + khf * 8],
                           *(const v8bf*)&pp[lm * 72 + kc * 32 + 16 + khf * 8]);
#pragma unroll
      for (int dt = 0; dt < 8; ++dt) {
        v16bf vb;
#pragma unroll
        for (int s = 0; s < 16; ++s)
          vb[s] = lsV[cur][(kc * 32 + (s >> 3) * 16 + khf * 8 + (s & 7)) * 136 + dt * 16 + lm];
        acco[dt] = WMMA_BF16(pa, vb, acco[dt]);
      }
    }
    wait_async0();
    __syncthreads();
  }
#pragma unroll
  for (int i = 0; i < 8; ++i) {
    const float inv = 1.0f / lstat[i];
    const int row = q0 + khf * 8 + i;
    bf16_t* ob = ctx + (((size_t)b * S + row) * Hn + h) * DH;
#pragma unroll
    for (int dt = 0; dt < 8; ++dt)
      ob[dt * 16 + lm] = (bf16_t)(acco[dt][i] * inv);
  }
}

// ---------------------------------------------------------------------------
// Final LayerNorm (bf16 in, f32 out), one block per row
// ---------------------------------------------------------------------------
__global__ __launch_bounds__(256) void layernorm_kernel(
    const bf16_t* __restrict__ x, const float* __restrict__ g,
    const float* __restrict__ b, float* __restrict__ out)
{
  const int D = 1024;
  __shared__ float rs[256], rs2[256];
  const int row = blockIdx.x;
  const int t = threadIdx.x;
  const bf16_t* xr = x + (size_t)row * D;
  float vals[4];
  float s = 0.0f, s2 = 0.0f;
#pragma unroll
  for (int i = 0; i < 4; ++i) {
    const float v = (float)xr[t + i * 256];
    vals[i] = v; s += v; s2 += v * v;
  }
  rs[t] = s; rs2[t] = s2;
  __syncthreads();
  for (int off = 128; off > 0; off >>= 1) {
    if (t < off) { rs[t] += rs[t + off]; rs2[t] += rs2[t + off]; }
    __syncthreads();
  }
  const float mu = rs[0] * (1.0f / D);
  const float var = rs2[0] * (1.0f / D) - mu * mu;
  const float rstd = rsqrtf(var + 1e-6f);
#pragma unroll
  for (int i = 0; i < 4; ++i) {
    const int c = t + i * 256;
    out[(size_t)row * D + c] = (vals[i] - mu) * rstd * g[c] + b[c];
  }
}

// ---------------------------------------------------------------------------
// Host orchestration
// ---------------------------------------------------------------------------
extern "C" void kernel_launch(void* const* d_in, const int* in_sizes, int n_in,
                              void* d_out, int out_size, void* d_ws, size_t ws_size,
                              hipStream_t stream)
{
  (void)in_sizes; (void)n_in; (void)out_size; (void)ws_size;
  const int D = 1024;
  const size_t ACT = (size_t)8 * 1024 * 1024;   // B*S*D
  const size_t WMAT = (size_t)D * D;

  const float* inputs = (const float*)d_in[0];
  const float* ew     = (const float*)d_in[1];
  const float* act_w  = (const float*)d_in[2];
  const float* We     = (const float*)d_in[3];
  const float* be     = (const float*)d_in[4];
  const float* Wq     = (const float*)d_in[5];
  const float* Wk     = (const float*)d_in[6];
  const float* Wv     = (const float*)d_in[7];
  const float* Wo     = (const float*)d_in[8];
  const float* ln_g   = (const float*)d_in[9];
  const float* ln_b   = (const float*)d_in[10];
  float* out = (float*)d_out;

  char* wsp = (char*)d_ws;
  auto alloc = [&](size_t bytes) -> bf16_t* {
    char* p = wsp;
    wsp += (bytes + 255) & ~(size_t)255;
    return (bf16_t*)p;
  };
  bf16_t* node[7];
  for (int i = 0; i < 7; ++i) node[i] = alloc(ACT * 2);  // node[-1..5]
  bf16_t* qB  = alloc(ACT * 2);
  bf16_t* kB  = alloc(ACT * 2);
  bf16_t* vB  = alloc(ACT * 2);
  bf16_t* qhB = alloc(ACT * 2);
  bf16_t* khB = alloc(ACT * 2);
  bf16_t* vhB = alloc(ACT * 2);
  bf16_t* wc[7];
  for (int i = 0; i < 7; ++i) wc[i] = alloc(WMAT * 2);

  auto conv = [&](const float* src, bf16_t* dst, size_t n) {
    convert_f32_bf16_kernel<<<1024, 256, 0, stream>>>(src, dst, n);
  };
  auto gemm0 = [&](const bf16_t* A, const bf16_t* W, bf16_t* o) {
    gemm_bf16_kernel<0><<<512, 256, 0, stream>>>(A, W, nullptr, nullptr, nullptr, o,
                                                 8192, 1024, 1024);
  };
  auto gemm1 = [&](const bf16_t* A, const bf16_t* W, const float* bias,
                   const float* gate, bf16_t* o) {
    gemm_bf16_kernel<1><<<512, 256, 0, stream>>>(A, W, bias, gate, nullptr, o,
                                                 8192, 1024, 1024);
  };
  auto gemm2 = [&](const bf16_t* A, const bf16_t* W, const float* gate,
                   const bf16_t* add, bf16_t* o) {
    gemm_bf16_kernel<2><<<512, 256, 0, stream>>>(A, W, nullptr, gate, add, o,
                                                 8192, 1024, 1024);
  };

  conv(inputs, node[0], ACT);

  int lind = 0;
  for (int c = 0; c < 6; ++c) {
    const int nsrc = (c + 1 < 5) ? (c + 1) : 5;
    const int snode = c - nsrc;
    const int eq = lind + nsrc - 1, ek = lind, ev = lind + nsrc / 2;
    const int in_q = c - 1, in_k = snode, in_v = snode + nsrc / 2;

    conv(We + (size_t)eq * WMAT, wc[0], WMAT);
    conv(We + (size_t)ek * WMAT, wc[1], WMAT);
    conv(We + (size_t)ev * WMAT, wc[2], WMAT);
    conv(Wq + (size_t)c  * WMAT, wc[3], WMAT);
    conv(Wk + (size_t)c  * WMAT, wc[4], WMAT);
    conv(Wv + (size_t)c  * WMAT, wc[5], WMAT);
    conv(Wo + (size_t)c  * WMAT, wc[6], WMAT);

    gemm1(node[in_q + 1], wc[0], be + (size_t)eq * D, ew + c * 3 + 0, qB);
    gemm1(node[in_k + 1], wc[1], be + (size_t)ek * D, ew + c * 3 + 1, kB);
    gemm1(node[in_v + 1], wc[2], be + (size_t)ev * D, ew + c * 3 + 2, vB);
    gemm0(qB, wc[3], qhB);
    gemm0(kB, wc[4], khB);
    gemm0(vB, wc[5], vhB);

    transpose_sdh_kernel<<<8192, 256, 0, stream>>>(khB, kB);  // kB := K^T [B,H,DH,S]
    attn_kernel<<<512, 256, 0, stream>>>(qhB, kB, vhB, khB);  // khB := ctx

    // node[c] = act_w[c] * (ctx @ Wo) + q
    gemm2(khB, wc[6], act_w + c, qB, node[c + 1]);
    lind += nsrc;
  }

  layernorm_kernel<<<8192, 256, 0, stream>>>(node[6], ln_g, ln_b, out);
}
